// EGTConv_74208444940579
// MI455X (gfx1250) — matrix operand order repaired and checked by reading, not compiled, not run on hardware
//
#include <hip/hip_runtime.h>
#include <hip/hip_bf16.h>

// EGT layer for MI455X (gfx1250, wave32, WMMA f32_16x16x32_f16).
// B=16 N=256 C=512 H=8 E=64 HID=64.

#define BB 16
#define NN 256
#define CC 512
#define HH 8
#define EE 64
#define HID 64
#define BNROWS (BB*NN)   // 4096
#define BH (BB*HH)       // 128

typedef __attribute__((ext_vector_type(16))) _Float16 v16h;
typedef __attribute__((ext_vector_type(8)))  float    v8f;

// ---------------- WMMA fragment helpers (wave32 layouts, ISA 7.12.2) ---------
// A 16x32 f16: lane l (l<16 -> half 0, else half 1), row = l&15.
// slots 0..7  -> k = 8*half + 0..7 ; slots 8..15 -> k = 16 + 8*half + 0..7
__device__ __forceinline__ v16h frag_a(const _Float16* src, int ld) {
  int lane = (int)(threadIdx.x & 31);
  int hf = lane >> 4, r = lane & 15;
  const _Float16* p = src + r * ld + hf * 8;
  v16h f;
#pragma unroll
  for (int s = 0; s < 8; ++s) { f[s] = p[s]; f[8 + s] = p[16 + s]; }
  return f;
}
// B 32x16 f16 given as B^T row-major ([n][k], k contiguous):
// lane = column n (l&15); slots 0..7 -> k = 8*half + 0..7 ; slots 8..15 -> k = 16 + 8*half + 0..7
__device__ __forceinline__ v16h frag_b(const _Float16* srcT, int ld) {
  int lane = (int)(threadIdx.x & 31);
  int hf = lane >> 4, n = lane & 15;
  const _Float16* p = srcT + n * ld + hf * 8;
  v16h f;
#pragma unroll
  for (int s = 0; s < 8; ++s) { f[s] = p[s]; f[8 + s] = p[16 + s]; }
  return f;
}
#define WMMA_F16(a, b, c) \
  __builtin_amdgcn_wmma_f32_16x16x32_f16(false, (a), false, (b), (short)0, (c), false, false)

// ---------------- weight prep: fp32 [K][N] -> f16 [N][K] ---------------------
__global__ void wprep_kernel(const float* src, _Float16* dst, int K, int N) {
  int idx = blockIdx.x * blockDim.x + threadIdx.x;
  if (idx >= K * N) return;
  int n = idx / K, k = idx % K;
  dst[idx] = (_Float16)src[(size_t)k * N + n];
}

// ---------------- LayerNorm over 512, fp32 in -> f16 out ---------------------
__global__ void ln512_kernel(const float* __restrict__ src, _Float16* __restrict__ dst,
                             const float* g, const float* b) {
  __shared__ float red[256];
  int row = blockIdx.x, t = threadIdx.x;
  const float* p = src + (size_t)row * CC;
  float x0 = p[t], x1 = p[t + 256];
  red[t] = x0 + x1; __syncthreads();
  for (int off = 128; off; off >>= 1) { if (t < off) red[t] += red[t + off]; __syncthreads(); }
  float mu = red[0] * (1.f / 512.f); __syncthreads();
  float d0 = x0 - mu, d1 = x1 - mu;
  red[t] = d0 * d0 + d1 * d1; __syncthreads();
  for (int off = 128; off; off >>= 1) { if (t < off) red[t] += red[t + off]; __syncthreads(); }
  float rs = rsqrtf(red[0] * (1.f / 512.f) + 1e-5f);
  dst[(size_t)row * CC + t]       = (_Float16)(d0 * rs * g[t] + b[t]);
  dst[(size_t)row * CC + t + 256] = (_Float16)(d1 * rs * g[t + 256] + b[t + 256]);
}

// ------------- fused edge LN + e_bias/gate projections (streaming) -----------
// one 64-thread block per edge; writes ebias into attn_hat buffer (B,H,N,N)
__global__ void edge_bias_gate_kernel(const float* __restrict__ edge_attr,
                                      const float* g, const float* bb,
                                      const float* W_ei, const float* b_ei,
                                      const float* W_gate, const float* b_gate,
                                      float* __restrict__ ebias, _Float16* __restrict__ gates) {
  int eidx = blockIdx.x;                 // b*65536 + l*256 + m
  int t = threadIdx.x;                   // 0..63
  int b = eidx >> 16, l = (eidx >> 8) & 255, m = eidx & 255;
  const float* p = edge_attr + (size_t)eidx * EE;
  __shared__ float red[64];
  __shared__ float eln[64];
  float v = p[t];
  red[t] = v; __syncthreads();
  for (int off = 32; off; off >>= 1) { if (t < off) red[t] += red[t + off]; __syncthreads(); }
  float mu = red[0] * (1.f / 64.f); __syncthreads();
  float d = v - mu;
  red[t] = d * d; __syncthreads();
  for (int off = 32; off; off >>= 1) { if (t < off) red[t] += red[t + off]; __syncthreads(); }
  float rs = rsqrtf(red[0] * (1.f / 64.f) + 1e-5f);
  eln[t] = d * rs * g[t] + bb[t];
  __syncthreads();
  if (t < 16) {
    int h = t & 7;
    const float* Wm = (t < 8) ? W_ei : W_gate;
    float s = 0.f;
    for (int e = 0; e < 64; ++e) s += eln[e] * Wm[e * 8 + h];
    size_t oidx = ((size_t)(b * 8 + h) * NN + l) * NN + m;
    if (t < 8) ebias[oidx] = s + b_ei[h];
    else       gates[oidx] = (_Float16)(1.f / (1.f + expf(-(s + b_gate[h]))));
  }
}

// ------------- qkv GEMM: hln(4096x512) @ WqkvT -> q/k/vT f16 -----------------
__global__ void gemm_qkv_kernel(const _Float16* __restrict__ A, const _Float16* __restrict__ BT,
                                const float* bias, _Float16* q, _Float16* k, _Float16* vT) {
  int nb = blockIdx.x, mb = blockIdx.y;
  int lane = threadIdx.x;
  const _Float16* Ap = A + (size_t)mb * 16 * CC;
  const _Float16* Bp = BT + (size_t)nb * 16 * CC;
  v8f c = {};
  for (int kk = 0; kk < CC; kk += 32)
    c = WMMA_F16(frag_a(Ap + kk, CC), frag_b(Bp + kk, CC), c);
  int col = nb * 16 + (lane & 15);
  int hf = lane >> 4;
  int dd = col >> 3, h = col & 7;
  float bv = bias[col];
#pragma unroll
  for (int i = 0; i < 8; ++i) {
    int m = mb * 16 + i + 8 * hf;
    int b = m >> 8, n = m & 255;
    _Float16 val = (_Float16)(c[i] + bv);
    if (dd < 64)        q[((size_t)(b * 8 + h) * NN + n) * HID + dd] = val;
    else if (dd < 128)  k[((size_t)(b * 8 + h) * NN + n) * HID + (dd - 64)] = val;
    else                vT[((size_t)(b * 8 + h) * HID + (dd - 128)) * NN + n] = val;
  }
}

// ------------- scores: per (b,h) QK^T, clip, += e_bias (in place) ------------
__global__ void gemm_scores_kernel(const _Float16* __restrict__ q, const _Float16* __restrict__ k,
                                   float* __restrict__ attn_hat) {
  int nb = blockIdx.x, mb = blockIdx.y, bh = blockIdx.z;
  int lane = threadIdx.x;
  const _Float16* Ap = q + (size_t)bh * NN * HID + (size_t)mb * 16 * HID;
  const _Float16* Bp = k + (size_t)bh * NN * HID + (size_t)nb * 16 * HID;
  v8f c = {};
  for (int kk = 0; kk < HID; kk += 32)
    c = WMMA_F16(frag_a(Ap + kk, HID), frag_b(Bp + kk, HID), c);
  int mcol = nb * 16 + (lane & 15);
  int hf = lane >> 4;
#pragma unroll
  for (int i = 0; i < 8; ++i) {
    int l = mb * 16 + i + 8 * hf;
    size_t idx = ((size_t)bh * NN + l) * NN + mcol;
    attn_hat[idx] = fminf(fmaxf(c[i], -5.f), 5.f) + attn_hat[idx];
  }
}

// ------------- softmax * gates + scaler (one 256-thread block per row) -------
__global__ void softmax_kernel(const float* __restrict__ attn_hat, const _Float16* __restrict__ gates,
                               _Float16* __restrict__ attn_tild, float* __restrict__ scaler) {
  int row = blockIdx.x;                  // (b*8+h)*256 + l
  int t = threadIdx.x;
  float a = attn_hat[(size_t)row * NN + t];
  float g = (float)gates[(size_t)row * NN + t];
  __shared__ float red[256];
  red[t] = a; __syncthreads();
  for (int off = 128; off; off >>= 1) { if (t < off) red[t] = fmaxf(red[t], red[t + off]); __syncthreads(); }
  float mx = red[0]; __syncthreads();
  float e = expf(a - mx);
  red[t] = e; __syncthreads();
  for (int off = 128; off; off >>= 1) { if (t < off) red[t] += red[t + off]; __syncthreads(); }
  float inv = 1.f / red[0]; __syncthreads();
  attn_tild[(size_t)row * NN + t] = (_Float16)(e * inv * g);
  red[t] = g; __syncthreads();
  for (int off = 128; off; off >>= 1) { if (t < off) red[t] += red[t + off]; __syncthreads(); }
  if (t == 0) {
    int l = row & 255;
    scaler[row] = (l < 1) ? 1.0f : log1pf(red[0]);   // V=1
  }
}

// ------------- v_attn = attn_tild @ V, scaled, scatter to (B,N,C) f16 --------
__global__ void gemm_av_kernel(const _Float16* __restrict__ attn_tild, const _Float16* __restrict__ vT,
                               const float* __restrict__ scaler, _Float16* __restrict__ v_attn) {
  int nb = blockIdx.x, mb = blockIdx.y, bh = blockIdx.z;
  int lane = threadIdx.x;
  const _Float16* Ap = attn_tild + (size_t)bh * NN * NN + (size_t)mb * 16 * NN;
  const _Float16* Bp = vT + (size_t)bh * HID * NN + (size_t)nb * 16 * NN;
  v8f c = {};
  for (int kk = 0; kk < NN; kk += 32)
    c = WMMA_F16(frag_a(Ap + kk, NN), frag_b(Bp + kk, NN), c);
  int d = nb * 16 + (lane & 15);
  int hf = lane >> 4;
  int b = bh >> 3, h = bh & 7;
#pragma unroll
  for (int i = 0; i < 8; ++i) {
    int l = mb * 16 + i + 8 * hf;
    float s = scaler[bh * NN + l];
    v_attn[((size_t)(b * NN + l)) * CC + d * 8 + h] = (_Float16)(c[i] * s);
  }
}

// ------------- 4096x512x512 GEMM + bias + fp32 residual -> fp32 --------------
__global__ void gemm512_resid_kernel(const _Float16* __restrict__ A, const _Float16* __restrict__ BT,
                                     const float* bias, const float* __restrict__ resid,
                                     float* __restrict__ out) {
  int nb = blockIdx.x, mb = blockIdx.y;
  int lane = threadIdx.x;
  const _Float16* Ap = A + (size_t)mb * 16 * CC;
  const _Float16* Bp = BT + (size_t)nb * 16 * CC;
  v8f c = {};
  for (int kk = 0; kk < CC; kk += 32)
    c = WMMA_F16(frag_a(Ap + kk, CC), frag_b(Bp + kk, CC), c);
  int col = nb * 16 + (lane & 15);
  int hf = lane >> 4;
  float bv = bias[col];
#pragma unroll
  for (int i = 0; i < 8; ++i) {
    int m = mb * 16 + i + 8 * hf;
    out[(size_t)m * CC + col] = c[i] + bv + resid[(size_t)m * CC + col];
  }
}

// ------------- 4096x512x512 GEMM + bias + ELU -> f16 -------------------------
__global__ void gemm512_elu_kernel(const _Float16* __restrict__ A, const _Float16* __restrict__ BT,
                                   const float* bias, _Float16* __restrict__ out) {
  int nb = blockIdx.x, mb = blockIdx.y;
  int lane = threadIdx.x;
  const _Float16* Ap = A + (size_t)mb * 16 * CC;
  const _Float16* Bp = BT + (size_t)nb * 16 * CC;
  v8f c = {};
  for (int kk = 0; kk < CC; kk += 32)
    c = WMMA_F16(frag_a(Ap + kk, CC), frag_b(Bp + kk, CC), c);
  int col = nb * 16 + (lane & 15);
  int hf = lane >> 4;
  float bv = bias[col];
#pragma unroll
  for (int i = 0; i < 8; ++i) {
    int m = mb * 16 + i + 8 * hf;
    float x = c[i] + bv;
    out[(size_t)m * CC + col] = (_Float16)(x > 0.f ? x : expm1f(x));
  }
}

// ------------- fused edge output pipeline (1 wave per 16-edge tile) ----------
// out1_e = attn_hat@W_eo + b_eo + edge_attr; LN; elu FFN (WMMA 16x64x64 x2);
// out2_e written directly -- no 268MB intermediates.
__global__ void edge_out_kernel(const float* __restrict__ attn_hat, const float* __restrict__ edge_attr,
                                const float* W_eo, const float* b_eo,
                                const float* lng, const float* lnb,
                                const _Float16* __restrict__ Wef1T, const float* b_ef1,
                                const _Float16* __restrict__ Wef2T, const float* b_ef2,
                                float* __restrict__ out_e) {
  int tile = blockIdx.x;                // b*4096 + l*16 + mtile
  int lane = threadIdx.x;               // 0..31
  int b = tile >> 12, l = (tile >> 4) & 255, mb = (tile & 15) * 16;
  __shared__ float     ah[16][8];
  __shared__ float     ot[16][64];
  __shared__ _Float16  aln[16][64];
  __shared__ _Float16  tt[16][64];
#pragma unroll
  for (int qd = 0; qd < 4; ++qd) {
    int idx = lane * 4 + qd;            // 0..127
    int h = idx >> 4, i = idx & 15;
    ah[i][h] = attn_hat[(((size_t)(b * 8 + h) * NN + l) * NN) + mb + i];
  }
  __syncthreads();
  size_t ebase = (((size_t)(b * NN + l)) * NN + mb) * EE;
  for (int idx = lane; idx < 16 * 64; idx += 32) {
    int i = idx >> 6, e = idx & 63;
    float s = b_eo[e];
#pragma unroll
    for (int h = 0; h < 8; ++h) s += ah[i][h] * W_eo[h * 64 + e];
    ot[i][e] = s + edge_attr[ebase + (size_t)i * EE + e];
  }
  __syncthreads();
  if (lane < 16) {
    float mu = 0.f;
    for (int e = 0; e < 64; ++e) mu += ot[lane][e];
    mu *= (1.f / 64.f);
    float var = 0.f;
    for (int e = 0; e < 64; ++e) { float d = ot[lane][e] - mu; var += d * d; }
    float rs = rsqrtf(var * (1.f / 64.f) + 1e-5f);
    for (int e = 0; e < 64; ++e)
      aln[lane][e] = (_Float16)((ot[lane][e] - mu) * rs * lng[e] + lnb[e]);
  }
  __syncthreads();
  const _Float16* Ap = &aln[0][0];
#pragma unroll
  for (int nt = 0; nt < 4; ++nt) {
    v8f c = {};
    for (int kk = 0; kk < 64; kk += 32)
      c = WMMA_F16(frag_a(Ap + kk, 64), frag_b(Wef1T + nt * 16 * 64 + kk, 64), c);
    int e = nt * 16 + (lane & 15);
    int hf = lane >> 4;
    float bv = b_ef1[e];
#pragma unroll
    for (int i = 0; i < 8; ++i) {
      int r = i + 8 * hf;
      float x = c[i] + bv;
      tt[r][e] = (_Float16)(x > 0.f ? x : expm1f(x));
    }
  }
  __syncthreads();
  const _Float16* Tp = &tt[0][0];
#pragma unroll
  for (int nt = 0; nt < 4; ++nt) {
    v8f c = {};
    for (int kk = 0; kk < 64; kk += 32)
      c = WMMA_F16(frag_a(Tp + kk, 64), frag_b(Wef2T + nt * 16 * 64 + kk, 64), c);
    int e = nt * 16 + (lane & 15);
    int hf = lane >> 4;
    float bv = b_ef2[e];
#pragma unroll
    for (int i = 0; i < 8; ++i) {
      int r = i + 8 * hf;
      out_e[ebase + (size_t)r * EE + e] = c[i] + bv + ot[r][e];
    }
  }
}

// =============================================================================
extern "C" void kernel_launch(void* const* d_in, const int* in_sizes, int n_in,
                              void* d_out, int out_size, void* d_ws, size_t ws_size,
                              hipStream_t stream) {
  const float* x         = (const float*)d_in[0];
  const float* edge_attr = (const float*)d_in[1];
  const float* ln_h_g    = (const float*)d_in[3];
  const float* ln_h_b    = (const float*)d_in[4];
  const float* ln_e_g    = (const float*)d_in[5];
  const float* ln_e_b    = (const float*)d_in[6];
  const float* W_qkv     = (const float*)d_in[7];
  const float* b_qkv     = (const float*)d_in[8];
  const float* W_ei      = (const float*)d_in[9];
  const float* b_ei      = (const float*)d_in[10];
  const float* W_gate    = (const float*)d_in[11];
  const float* b_gate    = (const float*)d_in[12];
  const float* W_no      = (const float*)d_in[13];
  const float* b_no      = (const float*)d_in[14];
  const float* ffn_ln_g  = (const float*)d_in[15];
  const float* ffn_ln_b  = (const float*)d_in[16];
  const float* W_f1      = (const float*)d_in[17];
  const float* b_f1      = (const float*)d_in[18];
  const float* W_f2      = (const float*)d_in[19];
  const float* b_f2      = (const float*)d_in[20];
  const float* W_eo      = (const float*)d_in[21];
  const float* b_eo      = (const float*)d_in[22];
  const float* effn_ln_g = (const float*)d_in[23];
  const float* effn_ln_b = (const float*)d_in[24];
  const float* W_ef1     = (const float*)d_in[25];
  const float* b_ef1     = (const float*)d_in[26];
  const float* W_ef2     = (const float*)d_in[27];
  const float* b_ef2     = (const float*)d_in[28];

  float* out_h = (float*)d_out;                       // 4096*512
  float* out_e = out_h + (size_t)BNROWS * CC;         // 16*256*256*64

  // ---- workspace carve-up ----
  char* W = (char*)d_ws;
  size_t off = 0;
  auto take = [&](size_t bytes) { size_t o = off; off += (bytes + 255) & ~(size_t)255; return o; };
  _Float16* hln     = (_Float16*)(W + take((size_t)BNROWS * CC * 2));
  _Float16* WqkvT   = (_Float16*)(W + take((size_t)3 * CC * CC * 2));
  _Float16* WnoT    = (_Float16*)(W + take((size_t)CC * CC * 2));
  _Float16* Wf1T    = (_Float16*)(W + take((size_t)CC * CC * 2));
  _Float16* Wf2T    = (_Float16*)(W + take((size_t)CC * CC * 2));
  _Float16* Wef1T   = (_Float16*)(W + take((size_t)EE * EE * 2));
  _Float16* Wef2T   = (_Float16*)(W + take((size_t)EE * EE * 2));
  _Float16* qf      = (_Float16*)(W + take((size_t)BH * NN * HID * 2));
  _Float16* kf      = (_Float16*)(W + take((size_t)BH * NN * HID * 2));
  _Float16* vTf     = (_Float16*)(W + take((size_t)BH * HID * NN * 2));
  float*    attnhat = (float*)   (W + take((size_t)BH * NN * NN * 4));
  _Float16* gates   = (_Float16*)(W + take((size_t)BH * NN * NN * 2));
  _Float16* atild   = (_Float16*)(W + take((size_t)BH * NN * NN * 2));
  float*    scaler  = (float*)   (W + take((size_t)BH * NN * 4));
  _Float16* vattn   = (_Float16*)(W + take((size_t)BNROWS * CC * 2));
  float*    out1h   = (float*)   (W + take((size_t)BNROWS * CC * 4));
  _Float16* ffin    = (_Float16*)(W + take((size_t)BNROWS * CC * 2));
  _Float16* t1      = (_Float16*)(W + take((size_t)BNROWS * CC * 2));

  // ---- weight prep (f32 [K][N] -> f16 [N][K]) ----
  wprep_kernel<<<(CC * 3 * CC + 255) / 256, 256, 0, stream>>>(W_qkv, WqkvT, CC, 3 * CC);
  wprep_kernel<<<(CC * CC + 255) / 256, 256, 0, stream>>>(W_no, WnoT, CC, CC);
  wprep_kernel<<<(CC * CC + 255) / 256, 256, 0, stream>>>(W_f1, Wf1T, CC, CC);
  wprep_kernel<<<(CC * CC + 255) / 256, 256, 0, stream>>>(W_f2, Wf2T, CC, CC);
  wprep_kernel<<<(EE * EE + 255) / 256, 256, 0, stream>>>(W_ef1, Wef1T, EE, EE);
  wprep_kernel<<<(EE * EE + 255) / 256, 256, 0, stream>>>(W_ef2, Wef2T, EE, EE);

  // ---- node LN + edge LN/bias/gates (streaming, no e_ln materialized) ----
  ln512_kernel<<<BNROWS, 256, 0, stream>>>(x, hln, ln_h_g, ln_h_b);
  edge_bias_gate_kernel<<<BB * NN * NN, 64, 0, stream>>>(edge_attr, ln_e_g, ln_e_b,
                                                         W_ei, b_ei, W_gate, b_gate,
                                                         attnhat, gates);
  // ---- qkv ----
  gemm_qkv_kernel<<<dim3(3 * CC / 16, BNROWS / 16), 32, 0, stream>>>(hln, WqkvT, b_qkv, qf, kf, vTf);
  // ---- attention ----
  gemm_scores_kernel<<<dim3(NN / 16, NN / 16, BH), 32, 0, stream>>>(qf, kf, attnhat);
  softmax_kernel<<<BH * NN, 256, 0, stream>>>(attnhat, gates, atild, scaler);
  gemm_av_kernel<<<dim3(HID / 16, NN / 16, BH), 32, 0, stream>>>(atild, vTf, scaler, vattn);
  // ---- node output path ----
  gemm512_resid_kernel<<<dim3(CC / 16, BNROWS / 16), 32, 0, stream>>>(vattn, WnoT, b_no, x, out1h);
  ln512_kernel<<<BNROWS, 256, 0, stream>>>(out1h, ffin, ffn_ln_g, ffn_ln_b);
  gemm512_elu_kernel<<<dim3(CC / 16, BNROWS / 16), 32, 0, stream>>>(ffin, Wf1T, b_f1, t1);
  gemm512_resid_kernel<<<dim3(CC / 16, BNROWS / 16), 32, 0, stream>>>(t1, Wf2T, b_f2, out1h, out_h);
  // ---- fused edge output path (read edge_attr once, write out2_e once) ----
  edge_out_kernel<<<BB * NN * (NN / 16), 32, 0, stream>>>(attnhat, edge_attr, W_eo, b_eo,
                                                          effn_ln_g, effn_ln_b,
                                                          Wef1T, b_ef1, Wef2T, b_ef2, out_e);
}
